// DGCNNClassifier_28621662061130
// MI455X (gfx1250) — matrix operand order repaired and checked by reading, not compiled, not run on hardware
//
#include <hip/hip_runtime.h>
#include <cstddef>

typedef float  v8f  __attribute__((ext_vector_type(8)));
typedef __bf16 v8bf __attribute__((ext_vector_type(8)));
typedef __bf16 v16bf __attribute__((ext_vector_type(16)));

union V16 { v16bf v; v8bf h[2]; };

#define EPSBN 1e-5f

// ---------------------------------------------------------------------------
// kNN: one wave of 32 threads = 32 query points. Neighbor features tiled
// through LDS; per-thread insertion-sorted top-20 (descending pd, like top_k).
// ---------------------------------------------------------------------------
template<int CIN>
__global__ __launch_bounds__(32)
void knn_kernel(const float* __restrict__ src, int sstride, int soff,
                int* __restrict__ idx_out)
{
    __shared__ float q[32][CIN];
    __shared__ float nt[64][CIN];
    __shared__ float nn[64];

    const int b    = blockIdx.y;
    const int p0   = blockIdx.x * 32;
    const int lane = threadIdx.x;

    for (int e = lane; e < 32 * CIN; e += 32) {
        int p = e / CIN, c = e % CIN;
        q[p][c] = src[((size_t)b * 2048 + p0 + p) * sstride + soff + c];
    }
    __syncthreads();

    float qn = 0.f;
    for (int c = 0; c < CIN; ++c) qn += q[lane][c] * q[lane][c];

    float bestd[20];
    int   besti[20];
    for (int j = 0; j < 20; ++j) { bestd[j] = -3.0e38f; besti[j] = 0; }

    for (int t0 = 0; t0 < 2048; t0 += 64) {
        __syncthreads();
        for (int e = lane; e < 64 * CIN; e += 32) {
            int p = e / CIN, c = e % CIN;
            nt[p][c] = src[((size_t)b * 2048 + t0 + p) * sstride + soff + c];
        }
        for (int p = lane; p < 64; p += 32) {
            float s = 0.f;
            for (int c = 0; c < CIN; ++c) s += nt[p][c] * nt[p][c];
            nn[p] = s;
        }
        __syncthreads();
        for (int j = 0; j < 64; ++j) {
            float inner = 0.f;
            for (int c = 0; c < CIN; ++c) inner += q[lane][c] * nt[j][c];
            float d = 2.f * inner - qn - nn[j];
            if (d > bestd[19]) {
                int pos = 19;
                while (pos > 0 && bestd[pos - 1] < d) {
                    bestd[pos] = bestd[pos - 1];
                    besti[pos] = besti[pos - 1];
                    --pos;
                }
                bestd[pos] = d;
                besti[pos] = t0 + j;
            }
        }
    }
    for (int j = 0; j < 20; ++j)
        idx_out[((size_t)b * 2048 + p0 + lane) * 20 + j] = besti[j];
}

// ---------------------------------------------------------------------------
// Fused edge-conv: gather (x_j - x_i || x_i) into LDS bf16 A, WMMA against
// w^T, BN + LeakyReLU, max over k=20 neighbors. 4 points/WG -> 80 A-rows
// = 5 M-tiles, one per wave (160 threads).
// ---------------------------------------------------------------------------
template<int CIN, int COUT>
__global__ __launch_bounds__(160)
void edge_conv_kernel(const float* __restrict__ src, int sstride, int soff,
                      const float* __restrict__ w,
                      const float* __restrict__ gamma, const float* __restrict__ beta,
                      const float* __restrict__ mean,  const float* __restrict__ var,
                      const int* __restrict__ idx,
                      float* __restrict__ dst, int dstride, int doff)
{
    constexpr int K2 = (2 * CIN) < 32 ? 32 : (2 * CIN);

    alignas(32) __shared__ __bf16 As[80][K2];
    alignas(32) __shared__ __bf16 Ws[16][K2];
    __shared__ float stage[80][16];
    __shared__ float cent[4][CIN];
    __shared__ int   nidx[80];

    const int tid  = threadIdx.x;
    const int lane = tid & 31;
    const int wave = tid >> 5;
    const int hsel = lane >> 4;
    const int nl   = lane & 15;
    const int b    = blockIdx.y;
    const int p0   = blockIdx.x * 4;

    for (int e = tid; e < 4 * CIN; e += 160) {
        int pp = e / CIN, c = e % CIN;
        cent[pp][c] = src[((size_t)b * 2048 + p0 + pp) * sstride + soff + c];
    }
    for (int e = tid; e < 80; e += 160)
        nidx[e] = idx[((size_t)b * 2048 + p0 + e / 20) * 20 + e % 20];
    __syncthreads();

    for (int e = tid; e < 80 * K2; e += 160) {
        int r = e / K2, c = e % K2;
        int pp = r / 20;
        float val;
        if (c < CIN) {
            int nb = nidx[r];
            val = src[((size_t)b * 2048 + nb) * sstride + soff + c] - cent[pp][c];
        } else if (c < 2 * CIN) {
            val = cent[pp][c - CIN];
        } else {
            val = 0.f;
        }
        As[r][c] = (__bf16)val;
    }
    __syncthreads();

    const int mrow = wave * 16 + nl;

    for (int ntile = 0; ntile < COUT / 16; ++ntile) {
        for (int e = tid; e < 16 * K2; e += 160) {
            int o = e / K2, c = e % K2;
            Ws[o][c] = (c < 2 * CIN) ? (__bf16)w[(size_t)(ntile * 16 + o) * (2 * CIN) + c]
                                     : (__bf16)0.f;
        }
        __syncthreads();

        v8f acc = {};
        for (int kt = 0; kt < K2 / 32; ++kt) {
            V16 a, bb;
            a.v    = *(const v16bf*)&As[mrow][kt * 32 + 16 * hsel];
            bb.v   = *(const v16bf*)&Ws[nl][kt * 32 + 16 * hsel];
            acc = __builtin_amdgcn_wmma_f32_16x16x32_bf16(
                false, a.v, false, bb.v, (short)0, acc, false, false);
        }

        int ch = ntile * 16 + nl;
        float sc = gamma[ch] * rsqrtf(var[ch] + EPSBN);
        float sh = beta[ch] - mean[ch] * sc;
        for (int r = 0; r < 8; ++r) {
            float vv = acc[r] * sc + sh;
            vv = vv >= 0.f ? vv : 0.2f * vv;
            stage[wave * 16 + r + 8 * hsel][nl] = vv;
        }
        __syncthreads();

        if (tid < 64) {
            int pp = tid >> 4, c = tid & 15;
            float mx = -3.0e38f;
            for (int j = 0; j < 20; ++j) mx = fmaxf(mx, stage[pp * 20 + j][c]);
            dst[((size_t)b * 2048 + p0 + pp) * dstride + doff + ntile * 16 + c] = mx;
        }
        __syncthreads();
    }
}

// ---------------------------------------------------------------------------
// w5 layer: h = lrelu(bn(w5 @ xcat)), per-point GEMM 2048x512x1024 (per batch)
// ---------------------------------------------------------------------------
__global__ __launch_bounds__(64)
void gemm5_kernel(const float* __restrict__ xcat, const float* __restrict__ w5,
                  const float* __restrict__ gamma, const float* __restrict__ beta,
                  const float* __restrict__ mean,  const float* __restrict__ var,
                  float* __restrict__ h)
{
    alignas(32) __shared__ __bf16 Xs[32][512];
    alignas(32) __shared__ __bf16 Ws[16][512];

    const int tid  = threadIdx.x;
    const int lane = tid & 31;
    const int wave = tid >> 5;
    const int hsel = lane >> 4;
    const int nl   = lane & 15;
    const int b    = blockIdx.y;
    const int p0   = blockIdx.x * 32;

    for (int e = tid; e < 32 * 512; e += 64) {
        int p = e >> 9, c = e & 511;
        Xs[p][c] = (__bf16)xcat[((size_t)b * 2048 + p0 + p) * 512 + c];
    }

    const int mrow = wave * 16 + nl;

    for (int ntile = 0; ntile < 64; ++ntile) {
        for (int e = tid; e < 16 * 512; e += 64) {
            int o = e >> 9, c = e & 511;
            Ws[o][c] = (__bf16)w5[(size_t)(ntile * 16 + o) * 512 + c];
        }
        if (ntile + 1 < 64)
            __builtin_prefetch(&w5[(size_t)(ntile + 1) * 16 * 512], 0, 3);
        __syncthreads();

        v8f acc = {};
        for (int kt = 0; kt < 16; ++kt) {
            V16 a, bb;
            a.v    = *(const v16bf*)&Xs[mrow][kt * 32 + 16 * hsel];
            bb.v   = *(const v16bf*)&Ws[nl][kt * 32 + 16 * hsel];
            acc = __builtin_amdgcn_wmma_f32_16x16x32_bf16(
                false, a.v, false, bb.v, (short)0, acc, false, false);
        }

        int ch = ntile * 16 + nl;
        float sc = gamma[ch] * rsqrtf(var[ch] + EPSBN);
        float sh = beta[ch] - mean[ch] * sc;
        for (int r = 0; r < 8; ++r) {
            float vv = acc[r] * sc + sh;
            vv = vv >= 0.f ? vv : 0.2f * vv;
            int pt = p0 + wave * 16 + r + 8 * hsel;
            h[((size_t)b * 2048 + pt) * 1024 + ch] = vv;
        }
        __syncthreads();
    }
}

// ---------------------------------------------------------------------------
// Global max + mean pooling over points: g = [max_n h, mean_n h]  (8, 2048)
// ---------------------------------------------------------------------------
__global__ __launch_bounds__(128)
void reduce_kernel(const float* __restrict__ h, float* __restrict__ g)
{
    const int c = blockIdx.x * 128 + threadIdx.x;
    const int b = blockIdx.y;
    float mx = -3.0e38f, sm = 0.f;
    for (int n = 0; n < 2048; ++n) {
        float v = h[((size_t)b * 2048 + n) * 1024 + c];
        mx = fmaxf(mx, v);
        sm += v;
    }
    g[(size_t)b * 2048 + c]        = mx;
    g[(size_t)b * 2048 + 1024 + c] = sm * (1.f / 2048.f);
}

// ---------------------------------------------------------------------------
// Small FC layers (outputs <= 8x512): plain dot products.
// ---------------------------------------------------------------------------
__global__ __launch_bounds__(64)
void fc_kernel(const float* __restrict__ in, const float* __restrict__ w,
               const float* __restrict__ bias,
               const float* __restrict__ gamma, const float* __restrict__ beta,
               const float* __restrict__ mean,  const float* __restrict__ var,
               float* __restrict__ out, int IN, int OUT)
{
    const int o = blockIdx.x * 64 + threadIdx.x;
    const int b = blockIdx.y;
    if (o >= OUT) return;
    const float* ir = in + (size_t)b * IN;
    const float* wr = w + (size_t)o * IN;
    float s = 0.f;
    for (int i = 0; i < IN; ++i) s += ir[i] * wr[i];
    if (bias) s += bias[o];
    if (gamma) {
        float sc = gamma[o] * rsqrtf(var[o] + EPSBN);
        s = (s - mean[o]) * sc + beta[o];
        s = s >= 0.f ? s : 0.2f * s;
    }
    out[(size_t)b * OUT + o] = s;
}

// ---------------------------------------------------------------------------
extern "C" void kernel_launch(void* const* d_in, const int* in_sizes, int n_in,
                              void* d_out, int out_size, void* d_ws, size_t ws_size,
                              hipStream_t stream)
{
    (void)in_sizes; (void)n_in; (void)out_size; (void)ws_size;

    const float* x   = (const float*)d_in[0];
    const float* w1  = (const float*)d_in[1];
    const float* g1  = (const float*)d_in[2];
    const float* b1  = (const float*)d_in[3];
    const float* m1  = (const float*)d_in[4];
    const float* v1  = (const float*)d_in[5];
    const float* w2  = (const float*)d_in[6];
    const float* g2  = (const float*)d_in[7];
    const float* b2  = (const float*)d_in[8];
    const float* m2  = (const float*)d_in[9];
    const float* v2  = (const float*)d_in[10];
    const float* w3  = (const float*)d_in[11];
    const float* g3  = (const float*)d_in[12];
    const float* b3  = (const float*)d_in[13];
    const float* m3  = (const float*)d_in[14];
    const float* v3  = (const float*)d_in[15];
    const float* w4  = (const float*)d_in[16];
    const float* g4  = (const float*)d_in[17];
    const float* b4  = (const float*)d_in[18];
    const float* m4  = (const float*)d_in[19];
    const float* v4  = (const float*)d_in[20];
    const float* w5  = (const float*)d_in[21];
    const float* g5  = (const float*)d_in[22];
    const float* b5  = (const float*)d_in[23];
    const float* m5  = (const float*)d_in[24];
    const float* v5  = (const float*)d_in[25];
    const float* l1w = (const float*)d_in[26];
    const float* g6  = (const float*)d_in[27];
    const float* b6  = (const float*)d_in[28];
    const float* m6  = (const float*)d_in[29];
    const float* v6  = (const float*)d_in[30];
    const float* l2w = (const float*)d_in[31];
    const float* l2b = (const float*)d_in[32];
    const float* g7  = (const float*)d_in[33];
    const float* b7  = (const float*)d_in[34];
    const float* m7  = (const float*)d_in[35];
    const float* v7  = (const float*)d_in[36];
    const float* l3w = (const float*)d_in[37];
    const float* l3b = (const float*)d_in[38];

    // workspace layout
    float* xcat = (float*)d_ws;                                   // 8*2048*512 f32
    int*   idx  = (int*)(xcat + (size_t)8 * 2048 * 512);          // 8*2048*20
    float* h    = (float*)(idx + (size_t)8 * 2048 * 20);          // 8*2048*1024 f32
    float* g    = h + (size_t)8 * 2048 * 1024;                    // 8*2048
    float* f1   = g + (size_t)8 * 2048;                           // 8*512
    float* f2   = f1 + (size_t)8 * 512;                           // 8*256
    float* outf = (float*)d_out;                                  // 8*40

    const dim3 kb(32),  kg(64, 8);     // knn
    const dim3 eb(160), eg(512, 8);    // edge conv

    // block 1: knn on raw xyz (B,N,3), conv (6 -> 64), write xcat[:, :, 0:64)
    knn_kernel<3><<<kg, kb, 0, stream>>>(x, 3, 0, idx);
    edge_conv_kernel<3, 64><<<eg, eb, 0, stream>>>(
        x, 3, 0, w1, g1, b1, m1, v1, idx, xcat, 512, 0);

    // block 2: knn on x1, conv (128 -> 64), write xcat[:, :, 64:128)
    knn_kernel<64><<<kg, kb, 0, stream>>>(xcat, 512, 0, idx);
    edge_conv_kernel<64, 64><<<eg, eb, 0, stream>>>(
        xcat, 512, 0, w2, g2, b2, m2, v2, idx, xcat, 512, 64);

    // block 3: knn on x2, conv (128 -> 128), write xcat[:, :, 128:256)
    knn_kernel<64><<<kg, kb, 0, stream>>>(xcat, 512, 64, idx);
    edge_conv_kernel<64, 128><<<eg, eb, 0, stream>>>(
        xcat, 512, 64, w3, g3, b3, m3, v3, idx, xcat, 512, 128);

    // block 4: knn on x3, conv (256 -> 256), write xcat[:, :, 256:512)
    knn_kernel<128><<<kg, kb, 0, stream>>>(xcat, 512, 128, idx);
    edge_conv_kernel<128, 256><<<eg, eb, 0, stream>>>(
        xcat, 512, 128, w4, g4, b4, m4, v4, idx, xcat, 512, 256);

    // w5 (1024,512) GEMM over all points + bn + lrelu
    gemm5_kernel<<<dim3(64, 8), dim3(64), 0, stream>>>(xcat, w5, g5, b5, m5, v5, h);

    // global max+mean pooling -> g (8, 2048)
    reduce_kernel<<<dim3(8, 8), dim3(128), 0, stream>>>(h, g);

    // FC head
    fc_kernel<<<dim3(8, 8), dim3(64), 0, stream>>>(g,  l1w, nullptr, g6, b6, m6, v6, f1, 2048, 512);
    fc_kernel<<<dim3(4, 8), dim3(64), 0, stream>>>(f1, l2w, l2b,     g7, b7, m7, v7, f2, 512, 256);
    fc_kernel<<<dim3(1, 8), dim3(64), 0, stream>>>(f2, l3w, l3b, nullptr, nullptr, nullptr, nullptr,
                                                   outf, 256, 40);
}
